// MoE_23476291240453
// MI455X (gfx1250) — matrix operand order repaired and checked by reading, not compiled, not run on hardware
//
#include <hip/hip_runtime.h>
#include <cstdint>
#include <cstddef>

// ---- problem sizes (fixed by the reference) ----
#define T_TOK 8192      // B*S tokens
#define DDIM  512
#define HDIM  1024
#define NEXP  8
#define TOPK  2
#define NSLOT (T_TOK * TOPK)

typedef __attribute__((ext_vector_type(16))) __bf16 v16bf;
typedef __attribute__((ext_vector_type(8)))  float  v8f;

// ---- TDM availability / arity detection (device pass only) ----
#if defined(__HIP_DEVICE_COMPILE__) && defined(__has_builtin)
#  if __has_builtin(__builtin_amdgcn_tensor_load_to_lds)
#    define HAVE_TDM 1
#  endif
#endif
#ifndef HAVE_TDM
#  define HAVE_TDM 0
#endif
#if __has_include(<hip/amd_detail/amd_gfx1250_TDM.h>)
#  define TDM_SIX_ARGS 1   // therock-10.0 headers => clang-23 6-arg builtin
#else
#  define TDM_SIX_ARGS 0   // ROCm 7.2 clang-22 5-arg builtin
#endif

#if HAVE_TDM
typedef __attribute__((ext_vector_type(4))) unsigned int v4u;
typedef __attribute__((ext_vector_type(4))) int          v4i;
typedef __attribute__((ext_vector_type(8))) int          v8i;

// Issue one 2-D TDM tile load: tile0 (contiguous, elements) x tile1 (rows),
// 2-byte elements, row stride elemStride0 (elements), optional LDS padding of
// padAmountDW DWORDs after every (2<<padIntervalCode) DWORDs. D# layout per
// CDNA5 ISA 8.3/8.4. Tensor dims set huge: tiles are known in-bounds.
__device__ __forceinline__ void tdm_load_tile_2d(unsigned ldsOff, const void* gsrc,
                                                 unsigned tile0, unsigned tile1,
                                                 unsigned long long elemStride0,
                                                 unsigned padIntervalCode,
                                                 unsigned padAmountDW) {
  unsigned long long ga = (unsigned long long)(uintptr_t)gsrc;
  v4u g0;
  g0[0] = 1u;                                            // count=1, user descriptor
  g0[1] = ldsOff;                                        // lds_addr (bytes)
  g0[2] = (unsigned)ga;                                  // global_addr[31:0]
  g0[3] = (unsigned)((ga >> 32) & 0x01FFFFFFull) | (2u << 30);  // ga[56:32] | type=2
  const unsigned td0 = 0x40000000u, td1 = 0x40000000u;   // huge tensor dims
  unsigned w0 = (1u << 16);                              // data_size = 2 bytes
  if (padAmountDW)
    w0 |= (1u << 20) | (padIntervalCode << 22) | ((padAmountDW - 1u) << 25);
  v8i g1;
  g1[0] = (int)w0;                                       // mask=0,pad ctl
  g1[1] = (int)((td0 & 0xFFFFu) << 16);                  // [47:32]=0, [63:48]=td0.lo
  g1[2] = (int)(((td0 >> 16) & 0xFFFFu) | ((td1 & 0xFFFFu) << 16));
  g1[3] = (int)(((td1 >> 16) & 0xFFFFu) | ((tile0 & 0xFFFFu) << 16)); // tile_dim0
  g1[4] = (int)(tile1 & 0xFFFFu);                        // tile_dim1, tile_dim2=0
  g1[5] = (int)(unsigned)(elemStride0 & 0xFFFFFFFFull);  // dim0_stride lo32
  g1[6] = (int)(unsigned)((elemStride0 >> 32) & 0xFFFFull); // dim0_stride hi16
  g1[7] = 0;                                             // dim1_stride (2-D: unused)
  v4i z4 = {0, 0, 0, 0};
#if TDM_SIX_ARGS
  v8i z8 = {0, 0, 0, 0, 0, 0, 0, 0};
  __builtin_amdgcn_tensor_load_to_lds(g0, g1, z4, z4, z8, 0);
#else
  __builtin_amdgcn_tensor_load_to_lds(g0, g1, z4, z4, 0);
#endif
}
#endif  // HAVE_TDM

__device__ __forceinline__ unsigned short f2bf(float f) {
  unsigned u = __float_as_uint(f);
  unsigned r = u + 0x7FFFu + ((u >> 16) & 1u);   // round-to-nearest-even
  return (unsigned short)(r >> 16);
}

// ---------------- tiny utility kernels ----------------
__global__ void zero_f32(float* __restrict__ p, int n) {
  int i = blockIdx.x * blockDim.x + threadIdx.x;
  if (i < n) p[i] = 0.0f;
}

__global__ void init_counts(int* __restrict__ counts) {
  int i = threadIdx.x;
  if (i < NEXP) counts[i] = 0;
}

__global__ void cvt_bf16(const float* __restrict__ src, unsigned short* __restrict__ dst, int n) {
  int i = (blockIdx.x * blockDim.x + threadIdx.x) * 4;
  if (i + 3 < n) {
    float4 f = *(const float4*)(src + i);
    dst[i + 0] = f2bf(f.x); dst[i + 1] = f2bf(f.y);
    dst[i + 2] = f2bf(f.z); dst[i + 3] = f2bf(f.w);
  } else {
    for (; i < n; ++i) dst[i] = f2bf(src[i]);
  }
}

// ---------------- router: scores -> softmax -> top-2 ----------------
__global__ void router_kernel(const float* __restrict__ x,
                              const float* __restrict__ rw,   // [D,E]
                              const float* __restrict__ rb,   // [E]
                              int* __restrict__ tki, float* __restrict__ tkp,
                              int* __restrict__ counts) {
  int t = blockIdx.x * blockDim.x + threadIdx.x;
  if (t >= T_TOK) return;
  float s[NEXP];
  #pragma unroll
  for (int e = 0; e < NEXP; ++e) s[e] = rb[e];
  const float* xr = x + (size_t)t * DDIM;
  for (int d = 0; d < DDIM; ++d) {
    float xv = xr[d];
    const float* w = rw + (size_t)d * NEXP;
    #pragma unroll
    for (int e = 0; e < NEXP; ++e) s[e] += xv * w[e];
  }
  float m = s[0];
  #pragma unroll
  for (int e = 1; e < NEXP; ++e) m = fmaxf(m, s[e]);
  float p[NEXP], sum = 0.0f;
  #pragma unroll
  for (int e = 0; e < NEXP; ++e) { p[e] = __expf(s[e] - m); sum += p[e]; }
  float inv = 1.0f / sum;
  #pragma unroll
  for (int e = 0; e < NEXP; ++e) p[e] *= inv;
  int i0 = 0;
  #pragma unroll
  for (int e = 1; e < NEXP; ++e) if (p[e] > p[i0]) i0 = e;
  int i1 = (i0 == 0) ? 1 : 0;
  #pragma unroll
  for (int e = 0; e < NEXP; ++e) if (e != i0 && p[e] > p[i1]) i1 = e;
  tki[t * TOPK + 0] = i0; tkp[t * TOPK + 0] = p[i0];
  tki[t * TOPK + 1] = i1; tkp[t * TOPK + 1] = p[i1];
  atomicAdd(&counts[i0], 1);
  atomicAdd(&counts[i1], 1);
}

__global__ void scan_kernel(const int* __restrict__ counts,
                            int* __restrict__ offs, int* __restrict__ cursors) {
  if (threadIdx.x == 0 && blockIdx.x == 0) {
    int a = 0;
    for (int e = 0; e < NEXP; ++e) { offs[e] = a; cursors[e] = a; a += counts[e]; }
    offs[NEXP] = a;
  }
}

__global__ void scatter_kernel(const int* __restrict__ tki, const float* __restrict__ tkp,
                               int* __restrict__ cursors,
                               int* __restrict__ tokList, float* __restrict__ gateList) {
  int t = blockIdx.x * blockDim.x + threadIdx.x;
  if (t >= T_TOK) return;
  #pragma unroll
  for (int s = 0; s < TOPK; ++s) {
    int e = tki[t * TOPK + s];
    int pos = atomicAdd(&cursors[e], 1);
    tokList[pos] = t;
    gateList[pos] = tkp[t * TOPK + s];
  }
}

// ---------------- persistent WMMA expert GEMM ----------------
// MODE 1: h = gather(x) @ w1 + b1                 (A gathered via token_list)
// MODE 2: gated = silu(h@wg+bg) * (h@wv+bv)       (dual-B, shared A frag; A via TDM)
// MODE 3: out[t] += gate * (gated @ w2 + b2)      (f32 atomic combine; A via TDM)
template<int MODE>
__global__ __launch_bounds__(256)
void moe_gemm(const unsigned short* __restrict__ A,
              const unsigned short* __restrict__ B,
              const unsigned short* __restrict__ B2,
              const float* __restrict__ bias1,
              const float* __restrict__ bias2,
              unsigned short* __restrict__ outBf,
              float* __restrict__ outF,
              const int* __restrict__ offs,
              const int* __restrict__ tokList,
              const float* __restrict__ gateList,
              int KD, int ND, int ldA) {
  constexpr int BM = 128, BN = 128, BK = 32;
  __shared__ unsigned short As[BM][BK + 8];                      // 128x40 (80B rows: 16B-aligned)
  __shared__ unsigned short Bst[BN][BK + 4];                     // transposed: [n][k]
  __shared__ unsigned short Bst2[(MODE == 2) ? BN : 1][BK + 4];

  const int tid = threadIdx.x;
  const int wave = tid >> 5, lane = tid & 31, lo = lane & 15, hi = lane >> 4;
  const int nT = ND / BN;
  const v8f VZERO = {0.f, 0.f, 0.f, 0.f, 0.f, 0.f, 0.f, 0.f};
#if HAVE_TDM
  // low 32 bits of a generic LDS pointer == byte offset within the LDS aperture
  const unsigned asLds = (unsigned)(uintptr_t)(void*)&As[0][0];
#endif

  for (int work = blockIdx.x;; work += gridDim.x) {
    // map flat work index -> (expert, m-tile, n-tile) using device-side counts
    int rem = work, e = 0; bool found = false;
    for (e = 0; e < NEXP; ++e) {
      int cnt = offs[e + 1] - offs[e];
      int tiles = ((cnt + BM - 1) / BM) * nT;
      if (rem < tiles) { found = true; break; }
      rem -= tiles;
    }
    if (!found) break;
    const int mt = rem / nT, nt = rem % nT;
    const int sBase = offs[e] + mt * BM;
    const int rowEnd = offs[e + 1];
    const int nBase = nt * BN;
    const unsigned short* Bexp = B + (size_t)e * KD * ND;

    v8f acc[8], acc2[8];
    #pragma unroll
    for (int j = 0; j < 8; ++j) { acc[j] = VZERO; acc2[j] = VZERO; }

    // per-thread staging assignments (manual A path)
    const int ar = tid >> 1;            // A row 0..127 (2 threads/row)
    const int ah = (tid & 1) * 16;      // A col half
    int sRow = sBase + ar; if (sRow >= rowEnd) sRow = offs[e];
    const unsigned short* Arow =
        (MODE == 1) ? A + (size_t)tokList[sRow] * ldA
                    : A + (size_t)sRow * ldA;
    const int br = tid >> 3;            // B k-row 0..31
    const int bc = (tid & 7) * 16;      // B n-col chunk

    for (int k0 = 0; k0 < KD; k0 += BK) {
      // ---- stage A tile ----
#if HAVE_TDM
      if constexpr (MODE != 1) {
        // slot-major rows are contiguous: one TDM 2-D tile (BKxBM, stride ldA),
        // LDS pad 4 DWORDs after every 16 DWORDs reproduces As[][BK+8] layout.
        if (wave == 0) {
          tdm_load_tile_2d(asLds, A + (size_t)sBase * ldA + k0,
                           /*tile0=*/BK, /*tile1=*/BM,
                           (unsigned long long)ldA,
                           /*padIntervalCode 16DW=*/3u, /*padAmountDW=*/4u);
          __builtin_amdgcn_s_wait_tensorcnt(0);
        }
      } else
#endif
      {
        const uint4* src = (const uint4*)(Arow + k0 + ah);
        *(uint4*)&As[ar][ah + 0] = src[0];
        *(uint4*)&As[ar][ah + 8] = src[1];
      }
      // ---- stage B tile, transposed into Bst[n][k] ----
      {
        union { uint4 q[2]; unsigned short u[16]; } r;
        const unsigned short* g = Bexp + (size_t)(k0 + br) * ND + nBase + bc;
        r.q[0] = ((const uint4*)g)[0]; r.q[1] = ((const uint4*)g)[1];
        #pragma unroll
        for (int i = 0; i < 16; ++i) Bst[bc + i][br] = r.u[i];
        if constexpr (MODE == 2) {
          const unsigned short* g2 = B2 + (size_t)e * KD * ND + (size_t)(k0 + br) * ND + nBase + bc;
          r.q[0] = ((const uint4*)g2)[0]; r.q[1] = ((const uint4*)g2)[1];
          #pragma unroll
          for (int i = 0; i < 16; ++i) Bst2[bc + i][br] = r.u[i];
        }
        if (k0 + BK < KD)
          __builtin_prefetch(Bexp + (size_t)(k0 + BK + br) * ND + nBase + bc, 0, 1);
      }
      __syncthreads();

      // A fragment: 16x32 bf16, contiguous K (lanes 0-15: K 0-15; lanes 16-31: K 16-31)
      union Frag { v16bf v; unsigned d[8]; } af;
      {
        const unsigned short* arow = &As[wave * 16 + lo][0];
        #pragma unroll
        for (int v = 0; v < 8; ++v) {
          const int kk = 2 * v + hi * 16;
          af.d[v] = *(const unsigned*)&arow[kk];
        }
      }
      #pragma unroll
      for (int j = 0; j < 8; ++j) {
        Frag bf;
        const unsigned short* brow = &Bst[j * 16 + lo][0];
        #pragma unroll
        for (int v = 0; v < 8; ++v) bf.d[v] = *(const unsigned*)&brow[2 * v + hi * 16];
        acc[j] = __builtin_amdgcn_wmma_f32_16x16x32_bf16(
            false, af.v, false, bf.v, (short)0, acc[j], false, false);
        if constexpr (MODE == 2) {
          Frag bf2;
          const unsigned short* brow2 = &Bst2[j * 16 + lo][0];
          #pragma unroll
          for (int v = 0; v < 8; ++v) bf2.d[v] = *(const unsigned*)&brow2[2 * v + hi * 16];
          acc2[j] = __builtin_amdgcn_wmma_f32_16x16x32_bf16(
              false, af.v, false, bf2.v, (short)0, acc2[j], false, false);
        }
      }
      __syncthreads();
    }

    // epilogue: C/D layout — VGPR r: lanes 0-15 -> M=r, lanes 16-31 -> M=r+8
    #pragma unroll
    for (int j = 0; j < 8; ++j) {
      #pragma unroll
      for (int r = 0; r < 8; ++r) {
        const int s = sBase + wave * 16 + r + hi * 8;
        if (s >= rowEnd) continue;
        const int n = nBase + j * 16 + lo;
        if constexpr (MODE == 1) {
          float val = acc[j][r] + bias1[e * ND + n];
          outBf[(size_t)s * ND + n] = f2bf(val);
        } else if constexpr (MODE == 2) {
          float g  = acc[j][r]  + bias1[e * ND + n];
          float vv = acc2[j][r] + bias2[e * ND + n];
          float silu = g / (1.0f + __expf(-g));
          outBf[(size_t)s * ND + n] = f2bf(silu * vv);
        } else {
          float val = acc[j][r] + bias1[e * ND + n];
          atomicAdd(&outF[(size_t)tokList[s] * ND + n], val * gateList[s]);
        }
      }
    }
  }
}

// ---------------- host launcher ----------------
extern "C" void kernel_launch(void* const* d_in, const int* in_sizes, int n_in,
                              void* d_out, int out_size, void* d_ws, size_t ws_size,
                              hipStream_t stream) {
  const float* x        = (const float*)d_in[0];
  const float* router_w = (const float*)d_in[1];
  const float* router_b = (const float*)d_in[2];
  const float* w1       = (const float*)d_in[3];
  const float* b1       = (const float*)d_in[4];
  const float* wg       = (const float*)d_in[5];
  const float* bg       = (const float*)d_in[6];
  const float* wv       = (const float*)d_in[7];
  const float* bv       = (const float*)d_in[8];
  const float* w2       = (const float*)d_in[9];
  const float* b2       = (const float*)d_in[10];
  float* out = (float*)d_out;

  // carve workspace (~126 MB)
  char* p = (char*)d_ws;
  auto carve = [&](size_t bytes) -> char* {
    char* r = p; p += (bytes + 255) & ~(size_t)255; return r;
  };
  unsigned short* xb   = (unsigned short*)carve((size_t)T_TOK * DDIM * 2);
  unsigned short* w1b  = (unsigned short*)carve((size_t)NEXP * DDIM * HDIM * 2);
  unsigned short* wgb  = (unsigned short*)carve((size_t)NEXP * HDIM * HDIM * 2);
  unsigned short* wvb  = (unsigned short*)carve((size_t)NEXP * HDIM * HDIM * 2);
  unsigned short* w2b  = (unsigned short*)carve((size_t)NEXP * HDIM * DDIM * 2);
  unsigned short* hAct = (unsigned short*)carve((size_t)NSLOT * HDIM * 2);
  unsigned short* gAct = (unsigned short*)carve((size_t)NSLOT * HDIM * 2);
  int*   tki      = (int*)carve((size_t)NSLOT * 4);
  float* tkp      = (float*)carve((size_t)NSLOT * 4);
  int*   tokList  = (int*)carve((size_t)NSLOT * 4);
  float* gateList = (float*)carve((size_t)NSLOT * 4);
  int*   counts   = (int*)carve(NEXP * 4);
  int*   offs     = (int*)carve((NEXP + 1) * 4);
  int*   cursors  = (int*)carve(NEXP * 4);

  const int nOut = T_TOK * DDIM;
  zero_f32<<<(nOut + 255) / 256, 256, 0, stream>>>(out, nOut);
  init_counts<<<1, 32, 0, stream>>>(counts);

  auto cvt = [&](const float* s, unsigned short* d, int n) {
    cvt_bf16<<<(n / 4 + 255) / 256, 256, 0, stream>>>(s, d, n);
  };
  cvt(x,  xb,  T_TOK * DDIM);
  cvt(w1, w1b, NEXP * DDIM * HDIM);
  cvt(wg, wgb, NEXP * HDIM * HDIM);
  cvt(wv, wvb, NEXP * HDIM * HDIM);
  cvt(w2, w2b, NEXP * HDIM * DDIM);

  router_kernel<<<T_TOK / 256, 256, 0, stream>>>(x, router_w, router_b, tki, tkp, counts);
  scan_kernel<<<1, 32, 0, stream>>>(counts, offs, cursors);
  scatter_kernel<<<T_TOK / 256, 256, 0, stream>>>(tki, tkp, cursors, tokList, gateList);

  // stage 1: h = gather(x) @ w1 + b1
  moe_gemm<1><<<1024, 256, 0, stream>>>(xb, w1b, nullptr, b1, nullptr,
                                        hAct, nullptr, offs, tokList, nullptr,
                                        DDIM, HDIM, DDIM);
  // stage 2: gated = silu(h@wg+bg) * (h@wv+bv)
  moe_gemm<2><<<1024, 256, 0, stream>>>(hAct, wgb, wvb, bg, bv,
                                        gAct, nullptr, offs, tokList, nullptr,
                                        HDIM, HDIM, HDIM);
  // stage 3: out[t] += gate * (gated @ w2 + b2)
  moe_gemm<3><<<512, 256, 0, stream>>>(gAct, w2b, nullptr, b2, nullptr,
                                       nullptr, out, offs, tokList, gateList,
                                       HDIM, DDIM, HDIM);
}
